// MoE_72258529788657
// MI455X (gfx1250) — compile-verified, matrix operand order checked by reading, not run on hardware
//
#include <hip/hip_runtime.h>
#include <hip/hip_bf16.h>
#include <math.h>

// ---- problem constants (from reference setup_inputs) ----
#define T_TOK  2048
#define DMODEL 2048
#define NEXP   16
#define IDIM   1408
#define SHDIM  2816
#define TOPK   4

// LDS row strides (elements), padded to dodge bank conflicts, rows stay 16B aligned
#define XLD   (DMODEL + 8)
#define HLD_S (SHDIM + 8)
#define HLD_R (IDIM + 8)

typedef __attribute__((ext_vector_type(16))) __bf16 v16bf;
typedef __attribute__((ext_vector_type(8)))  float  v8f;

union FragU { v16bf v; unsigned int w[8]; };

__device__ __forceinline__ unsigned short f2bf(float f) {
  unsigned int b = __builtin_bit_cast(unsigned int, f);
  b += 0x7FFFu + ((b >> 16) & 1u);        // round-to-nearest-even
  return (unsigned short)(b >> 16);
}

// pack two fp32 into a packed bf16 pair with the CDNA5 single-instruction
// converter (no clang builtin on this toolchain -> inline asm)
__device__ __forceinline__ unsigned int pk2(float lo, float hi) {
  unsigned int r;
  asm("v_cvt_pk_bf16_f32 %0, %1, %2" : "=v"(r) : "v"(lo), "v"(hi));
  return r;
}

__device__ __forceinline__ float fast_rcp(float x) {
  return __builtin_amdgcn_rcpf(x);        // v_rcp_f32
}

__device__ __forceinline__ v8f wmma_bf16(v16bf a, v16bf b, v8f c) {
  return __builtin_amdgcn_wmma_f32_16x16x32_bf16(false, a, false, b, (short)0, c,
                                                 false, false);
}

// A-fragment (16x32 bf16, M x K) from an LDS tile, per ISA 7.12.2 layout:
// lane l: m = l&15, g = l>>4; v0..3 hold K = g*8..g*8+7, v4..7 hold K = 16+g*8..+7.
__device__ __forceinline__ v16bf load_frag_lds(const unsigned short* buf, int ld,
                                               int k0, int lane) {
  int m = lane & 15, g = lane >> 4;
  const unsigned short* p = buf + m * ld + k0 + g * 8;
  FragU f;
  *(uint4*)&f.w[0] = *(const uint4*)p;          // ds_load_b128
  *(uint4*)&f.w[4] = *(const uint4*)(p + 16);   // ds_load_b128
  return f.v;
}

// B-fragment (32x16 bf16, K x N) from a row-major fp32 weight matrix [N, K]:
// lane l owns column n = n0 + (l&15); same K striping as A (symmetric layout).
// Streams fp32 rows with global_load_b128 and packs to bf16 with v_cvt_pk_bf16_f32.
__device__ __forceinline__ v16bf load_frag_g32(const float* __restrict__ W,
                                               long long ldk, int n0, int k0,
                                               int lane) {
  int n = n0 + (lane & 15), g = lane >> 4;
  const float* p = W + (long long)n * ldk + k0 + g * 8;
  float4 a = *(const float4*)p;
  float4 b = *(const float4*)(p + 4);
  float4 c = *(const float4*)(p + 16);
  float4 d = *(const float4*)(p + 20);
  FragU f;
  f.w[0] = pk2(a.x, a.y); f.w[1] = pk2(a.z, a.w);
  f.w[2] = pk2(b.x, b.y); f.w[3] = pk2(b.z, b.w);
  f.w[4] = pk2(c.x, c.y); f.w[5] = pk2(c.z, c.w);
  f.w[6] = pk2(d.x, d.y); f.w[7] = pk2(d.z, d.w);
  return f.v;
}

// C/D layout (32-bit 16x16): lane l col n = n0+(l&15); VGPR r row m = r + 8*(l>>4).
// Apply SwiGLU gate (+ optional per-row routing weight) and store bf16 row-major H.
__device__ __forceinline__ void store_gated(unsigned short* H, int ld, int n0,
                                            int lane, v8f h1, v8f h3,
                                            const float* wt) {
  int n = n0 + (lane & 15), g = lane >> 4;
#pragma unroll
  for (int r = 0; r < 8; ++r) {
    int m = r + 8 * g;
    float a = h1[r];
    float s = a * fast_rcp(1.0f + __expf(-a));   // silu, no IEEE divide
    float w = wt ? wt[m] : 1.0f;
    H[m * ld + n] = f2bf(s * h3[r] * w);
  }
}

// ---------------- kernel 0: zero per-expert counters ----------------
__global__ void moe_zero_counts(int* __restrict__ counts) {
  if (threadIdx.x < NEXP) counts[threadIdx.x] = 0;
}

// ---------------- kernel 1: gate + per-expert token gather lists ----------------
__global__ void moe_gate_kernel(const float* __restrict__ x,
                                const float* __restrict__ gw,
                                int* __restrict__ counts,
                                int* __restrict__ tokids,
                                float* __restrict__ tokw) {
  __shared__ float part[256];
  __shared__ float sc[NEXP];
  const int t = blockIdx.x;
  const int tid = threadIdx.x;
  const int e = tid >> 4;          // 16 threads per expert
  const int l = tid & 15;
  const float* xr = x + (size_t)t * DMODEL;
  const float* wr = gw + (size_t)e * DMODEL;
  float s = 0.f;
  for (int k = l; k < DMODEL; k += 16) s += xr[k] * wr[k];
  part[tid] = s;
  __syncthreads();
  if (tid < NEXP) {
    float acc = 0.f;
#pragma unroll
    for (int i = 0; i < 16; ++i) acc += part[tid * 16 + i];
    sc[tid] = fast_rcp(1.0f + __expf(-acc));    // sigmoid score
  }
  __syncthreads();
  if (tid == 0) {
    // group scores = amax over groups of 4; pick top-2 groups
    float gs[4];
#pragma unroll
    for (int g = 0; g < 4; ++g) {
      float m = sc[g * 4];
#pragma unroll
      for (int j = 1; j < 4; ++j) m = fmaxf(m, sc[g * 4 + j]);
      gs[g] = m;
    }
    int g0 = 0;
    for (int g = 1; g < 4; ++g) if (gs[g] > gs[g0]) g0 = g;
    int g1 = -1;
    for (int g = 0; g < 4; ++g) {
      if (g == g0) continue;
      if (g1 < 0 || gs[g] > gs[g1]) g1 = g;
    }
    // top-4 experts inside the two winning groups
    bool used[NEXP];
    for (int i = 0; i < NEXP; ++i) used[i] = false;
    int idx[TOPK]; float wv[TOPK]; float wsum = 0.f;
    for (int k = 0; k < TOPK; ++k) {
      int best = -1; float bv = -1.f;
      for (int i = 0; i < NEXP; ++i) {
        int gg = i >> 2;
        if (gg != g0 && gg != g1) continue;
        if (!used[i] && sc[i] > bv) { bv = sc[i]; best = i; }
      }
      used[best] = true; idx[k] = best; wv[k] = sc[best]; wsum += sc[best];
    }
    float inv = 1.0f / wsum;                    // ROUTE_SCALE == 1
    for (int k = 0; k < TOPK; ++k) {
      int ee = idx[k];
      int slot = atomicAdd(&counts[ee], 1);
      tokids[ee * T_TOK + slot] = t;
      tokw [ee * T_TOK + slot] = wv[k] * inv;
    }
  }
}

// vectorized fp32 -> bf16 row staging into LDS (float4 load, uint2 ds store)
__device__ __forceinline__ void stage_row_chunk(unsigned short* xb,
                                                const float* __restrict__ src,
                                                int m, int k) {
  float4 v = *(const float4*)(src + k);
  uint2 p;
  p.x = pk2(v.x, v.y);
  p.y = pk2(v.z, v.w);
  *(uint2*)(xb + m * XLD + k) = p;
}

// ---------------- kernel 2: shared SwiGLU MLP, writes z into d_out ----------------
__global__ void moe_shared_kernel(const float* __restrict__ x,
                                  const float* __restrict__ Ws1,
                                  const float* __restrict__ Ws2,
                                  const float* __restrict__ Ws3,
                                  float* __restrict__ out) {
  extern __shared__ unsigned char dynsmem[];
  unsigned short* xb = (unsigned short*)dynsmem;          // 16 x XLD bf16
  unsigned short* Hs = xb + 16 * XLD;                     // 16 x HLD_S bf16
  const int t0 = blockIdx.x * 16;
  const int tid = threadIdx.x;
  const int lane = tid & 31;
  const int wave = tid >> 5;

  // stage 16 token rows, fp32 -> bf16, into LDS (4 elements per iteration)
  for (int i = tid; i < 16 * DMODEL / 4; i += 256) {
    int m = i >> 9;                 // (i*4) / DMODEL
    int k = (i & 511) << 2;
    stage_row_chunk(xb, x + (size_t)(t0 + m) * DMODEL, m, k);
  }
  __syncthreads();

  // phase 1: H = silu(x Ws1^T) * (x Ws3^T)   [16 x 2816]
  for (int nt = wave; nt < SHDIM / 16; nt += 8) {
    int n0 = nt * 16;
    v8f a1 = {}, a3 = {};
    for (int k0 = 0; k0 < DMODEL; k0 += 32) {
      __builtin_prefetch(Ws1 + (size_t)(n0 + (lane & 15)) * DMODEL + k0 + 32, 0, 1);
      v16bf A  = load_frag_lds(xb, XLD, k0, lane);
      v16bf B1 = load_frag_g32(Ws1, DMODEL, n0, k0, lane);
      v16bf B3 = load_frag_g32(Ws3, DMODEL, n0, k0, lane);
      a1 = wmma_bf16(A, B1, a1);
      a3 = wmma_bf16(A, B3, a3);
    }
    store_gated(Hs, HLD_S, n0, lane, a1, a3, (const float*)0);
  }
  __syncthreads();

  // phase 2: z = H @ Ws2^T   (Ws2 is [D, SH] row-major), plain stores
  const int g = lane >> 4, nlo = lane & 15;
  for (int j = 0; j < 16; ++j) {
    int n0 = (wave * 16 + j) * 16;
    v8f acc = {};
    for (int k0 = 0; k0 < SHDIM; k0 += 32) {
      v16bf A = load_frag_lds(Hs, HLD_S, k0, lane);
      v16bf B = load_frag_g32(Ws2, SHDIM, n0, k0, lane);
      acc = wmma_bf16(A, B, acc);
    }
    int n = n0 + nlo;
#pragma unroll
    for (int r = 0; r < 8; ++r) {
      int m = r + 8 * g;
      out[(size_t)(t0 + m) * DMODEL + n] = acc[r];
    }
  }
}

// ---------------- kernel 3: routed experts with token gather, atomic scatter ----------------
__global__ void moe_routed_kernel(const float* __restrict__ x,
                                  const float* __restrict__ W1,
                                  const float* __restrict__ W2,
                                  const float* __restrict__ W3,
                                  const int* __restrict__ counts,
                                  const int* __restrict__ tokids,
                                  const float* __restrict__ tokw,
                                  float* __restrict__ out) {
  extern __shared__ unsigned char dynsmem[];
  unsigned short* xb = (unsigned short*)dynsmem;          // 16 x XLD bf16
  unsigned short* Hb = xb + 16 * XLD;                     // 16 x HLD_R bf16
  __shared__ int   rowtok[16];
  __shared__ float roww[16];

  const int e    = blockIdx.x >> 7;       // 128 tiles per expert (worst case)
  const int tile = blockIdx.x & 127;
  const int nt   = counts[e];
  if (tile * 16 >= nt) return;            // uniform early-exit: expert sparsity

  const int tid = threadIdx.x;
  const int lane = tid & 31;
  const int wave = tid >> 5;

  if (tid < 16) {
    int slot = tile * 16 + tid;
    if (slot < nt) {
      rowtok[tid] = tokids[e * T_TOK + slot];
      roww[tid]   = tokw [e * T_TOK + slot];
    } else {
      rowtok[tid] = 0;
      roww[tid]   = 0.0f;                 // zero weight -> zero contribution
    }
  }
  __syncthreads();

  // gather 16 routed token rows into LDS as bf16 (4 elements per iteration)
  for (int i = tid; i < 16 * DMODEL / 4; i += 256) {
    int m = i >> 9;
    int k = (i & 511) << 2;
    stage_row_chunk(xb, x + (size_t)rowtok[m] * DMODEL, m, k);
  }
  __syncthreads();

  const float* W1e = W1 + (size_t)e * IDIM * DMODEL;
  const float* W3e = W3 + (size_t)e * IDIM * DMODEL;
  const float* W2e = W2 + (size_t)e * DMODEL * IDIM;

  // phase 1: H = silu(x W1^T) * (x W3^T) * route_w   [16 x 1408]
  for (int ntl = wave; ntl < IDIM / 16; ntl += 8) {
    int n0 = ntl * 16;
    v8f a1 = {}, a3 = {};
    for (int k0 = 0; k0 < DMODEL; k0 += 32) {
      __builtin_prefetch(W1e + (size_t)(n0 + (lane & 15)) * DMODEL + k0 + 32, 0, 1);
      v16bf A  = load_frag_lds(xb, XLD, k0, lane);
      v16bf B1 = load_frag_g32(W1e, DMODEL, n0, k0, lane);
      v16bf B3 = load_frag_g32(W3e, DMODEL, n0, k0, lane);
      a1 = wmma_bf16(A, B1, a1);
      a3 = wmma_bf16(A, B3, a3);
    }
    store_gated(Hb, HLD_R, n0, lane, a1, a3, roww);
  }
  __syncthreads();

  // phase 2: y[tok] += H @ W2^T  (W2e is [D, I] row-major), scatter via f32 atomics
  const int g = lane >> 4, nlo = lane & 15;
  for (int j = 0; j < 16; ++j) {
    int n0 = (wave * 16 + j) * 16;
    v8f acc = {};
    for (int k0 = 0; k0 < IDIM; k0 += 32) {
      __builtin_prefetch(W2e + (size_t)(n0 + (lane & 15)) * IDIM + k0 + 32, 0, 1);
      v16bf A = load_frag_lds(Hb, HLD_R, k0, lane);
      v16bf B = load_frag_g32(W2e, IDIM, n0, k0, lane);
      acc = wmma_bf16(A, B, acc);
    }
    int n = n0 + nlo;
#pragma unroll
    for (int r = 0; r < 8; ++r) {
      int m = r + 8 * g;
      if (roww[m] != 0.0f)
        atomicAdd(&out[(size_t)rowtok[m] * DMODEL + n], acc[r]);
    }
  }
}

extern "C" void kernel_launch(void* const* d_in, const int* in_sizes, int n_in,
                              void* d_out, int out_size, void* d_ws, size_t ws_size,
                              hipStream_t stream) {
  const float* x      = (const float*)d_in[0];
  const float* gate_w = (const float*)d_in[1];
  const float* W1     = (const float*)d_in[2];
  const float* W2     = (const float*)d_in[3];
  const float* W3     = (const float*)d_in[4];
  const float* Ws1    = (const float*)d_in[5];
  const float* Ws2    = (const float*)d_in[6];
  const float* Ws3    = (const float*)d_in[7];
  float* out = (float*)d_out;

  // workspace layout: counts[E] | tokids[E*T] | tokw[E*T]   (~262 KB)
  int*   counts = (int*)d_ws;
  int*   tokids = counts + NEXP;
  float* tokw   = (float*)(tokids + NEXP * T_TOK);

  moe_zero_counts<<<1, 32, 0, stream>>>(counts);
  moe_gate_kernel<<<T_TOK, 256, 0, stream>>>(x, gate_w, counts, tokids, tokw);

  size_t sh_shared = (size_t)16 * XLD * 2 + (size_t)16 * HLD_S * 2;   // ~156 KB LDS
  moe_shared_kernel<<<T_TOK / 16, 256, sh_shared, stream>>>(x, Ws1, Ws2, Ws3, out);

  size_t sh_routed = (size_t)16 * XLD * 2 + (size_t)16 * HLD_R * 2;   // ~111 KB LDS
  moe_routed_kernel<<<NEXP * (T_TOK / 16), 256, sh_routed, stream>>>(
      x, W1, W2, W3, counts, tokids, tokw, out);
}
